// GraphConv2d_34368328302636
// MI455X (gfx1250) — matrix operand order repaired and checked by reading, not compiled, not run on hardware
//
#include <hip/hip_runtime.h>
#include <math.h>

typedef __attribute__((ext_vector_type(16))) _Float16 v16h;
typedef __attribute__((ext_vector_type(8)))  float    v8f;
typedef __attribute__((ext_vector_type(4)))  _Float16 h4;

union Frag {
    v16h v;
    uint4 q[2];
};

#define BB 4
#define CIN 192
#define COUT 384
#define NN 10000
#define KK 16
#define GG 4
#define BN_EPS 1e-5f

// ---------------------------------------------------------------------------
// Kernel 1: transpose x (B, CIN, N) f32 -> xt (B*N, CIN) f16, tiled via LDS
// ---------------------------------------------------------------------------
__global__ __launch_bounds__(256) void transpose_to_f16(
    const float* __restrict__ x, _Float16* __restrict__ xt)
{
    __shared__ float tile[32][33];
    const int b  = blockIdx.z;
    const int n0 = blockIdx.x * 32;
    const int c0 = blockIdx.y * 32;
    const int tx = threadIdx.x;   // 0..31
    const int ty = threadIdx.y;   // 0..7

    #pragma unroll
    for (int i = 0; i < 32; i += 8) {
        int c = c0 + ty + i;
        int n = n0 + tx;
        float v = 0.0f;
        if (c < CIN && n < NN) v = x[((size_t)b * CIN + c) * NN + n];
        tile[ty + i][tx] = v;
    }
    __syncthreads();
    #pragma unroll
    for (int i = 0; i < 32; i += 8) {
        int n = n0 + ty + i;
        int c = c0 + tx;
        if (c < CIN && n < NN)
            xt[((size_t)b * NN + n) * CIN + c] = (_Float16)tile[tx][ty + i];
    }
}

// ---------------------------------------------------------------------------
// Kernel 2: convert weights (COUT, 96) f32 -> f16
// ---------------------------------------------------------------------------
__global__ void wconv_kernel(const float* __restrict__ w, _Float16* __restrict__ wf, int n)
{
    int i = blockIdx.x * blockDim.x + threadIdx.x;
    if (i < n) wf[i] = (_Float16)w[i];
}

// ---------------------------------------------------------------------------
// Kernel 3: max-relative conv, 4 channels/thread (8B vector gathers),
// 4 nodes per 192-thread block. rel[b,n,c] = max_k( xt[b,j_k,c] - xt[b,i_k,c] )
// edge_index layout: (2, B, N, K) int32
// ---------------------------------------------------------------------------
#define NPB 4                 // nodes per block
#define CTH (CIN / 4)         // 48 channel-threads per node

__global__ __launch_bounds__(192) void mrconv_kernel(
    const _Float16* __restrict__ xt, const int* __restrict__ eidx,
    _Float16* __restrict__ rel)
{
    const int b     = blockIdx.y;
    const int node0 = blockIdx.x * NPB;
    const int tid   = threadIdx.x;       // 0..191
    const int sub   = tid / CTH;         // node within block, 0..3
    const int c0    = (tid % CTH) * 4;   // channel base

    __shared__ int sj[NPB][KK];
    __shared__ int si[NPB][KK];
    if (tid < NPB * KK) {
        int nn = tid / KK, k = tid % KK;
        sj[nn][k] = eidx[(((size_t)0 * BB + b) * NN + node0 + nn) * KK + k];
    } else if (tid < 2 * NPB * KK) {
        int t = tid - NPB * KK;
        int nn = t / KK, k = t % KK;
        si[nn][k] = eidx[(((size_t)1 * BB + b) * NN + node0 + nn) * KK + k];
    }
    __syncthreads();

    float m0 = -3.4e38f, m1 = -3.4e38f, m2 = -3.4e38f, m3 = -3.4e38f;
    #pragma unroll 4
    for (int k = 0; k < KK; ++k) {
        h4 vj = *(const h4*)(xt + ((size_t)b * NN + sj[sub][k]) * CIN + c0);
        h4 vi = *(const h4*)(xt + ((size_t)b * NN + si[sub][k]) * CIN + c0);
        m0 = fmaxf(m0, (float)vj[0] - (float)vi[0]);
        m1 = fmaxf(m1, (float)vj[1] - (float)vi[1]);
        m2 = fmaxf(m2, (float)vj[2] - (float)vi[2]);
        m3 = fmaxf(m3, (float)vj[3] - (float)vi[3]);
    }
    h4 r;
    r[0] = (_Float16)m0; r[1] = (_Float16)m1;
    r[2] = (_Float16)m2; r[3] = (_Float16)m3;
    *(h4*)(rel + ((size_t)b * NN + node0 + sub) * CIN + c0) = r;
}

// ---------------------------------------------------------------------------
// Kernel 4: grouped GEMM via v_wmma_f32_16x16x32_f16.
// One wave per 16x32 output tile (two 16-col WMMA tiles sharing A fragments).
// Groups 0,1 read xt; groups 2,3 read rel (each group's 96 inputs live
// entirely in one buffer). All fragments preloaded, then 6 WMMAs.
//
// A layout as loaded here (16-bit 16x32): M = lane&15 for both halves;
//   elem e -> K = half*16 + e
// B layout as loaded here (16-bit 32x16): col = lane&15; elem e -> K = half*16 + e
// D layout (f32 16x16): col = lane&15; vgpr r -> M = r + half*8
//   => each lane's 8 results are 8 consecutive n for one channel: 2x b128 store
// ---------------------------------------------------------------------------
#define WAVES_PER_BLOCK 8
#define M_TILES ((BB * NN) / 16)   // 2500
#define NT_PAIRS 3                 // 96/32 channel-pair tiles per group

__global__ __launch_bounds__(256) void gemm_wmma_kernel(
    const _Float16* __restrict__ xt,   // (B*N, 192)
    const _Float16* __restrict__ rel,  // (B*N, 192)
    const _Float16* __restrict__ wf,   // (384, 96)
    const float* __restrict__ bias,    // (384)
    float* __restrict__ out)           // (B, 384, N)
{
    const int lane   = threadIdx.x & 31;
    const int waveId = blockIdx.x * WAVES_PER_BLOCK + (threadIdx.x >> 5);
    // total waves = M_TILES * GG * NT_PAIRS = 30000 (exact for 3750 blocks)

    const int mt   = waveId % M_TILES;
    const int rest = waveId / M_TILES;      // 0..11
    const int g    = rest / NT_PAIRS;       // group 0..3
    const int ntp  = rest % NT_PAIRS;       // 32-channel pair tile

    const int row  = lane & 15;             // A row / B col / D col
    const int half = lane >> 4;
    const int sample0 = mt * 16;
    const int bb = sample0 / NN;            // wave-uniform (tiles never cross b)
    const int n0 = sample0 - bb * NN;

    const int colbase = (g & 1) * 96;
    const _Float16* __restrict__ src = (g < 2) ? xt : rel;
    const int o0 = g * 96 + ntp * 32 + row; // output channel, tile 0
    const int o1 = o0 + 16;                 // output channel, tile 1

    // ---- preload all fragments (6 x b128 A, 12 x pairs B) ----
    Frag a[3], bL[3], bH[3];
    const _Float16* ap  = src + (size_t)(sample0 + row) * 192 + colbase + half * 16;
    const _Float16* bp0 = wf + (size_t)o0 * 96 + half * 16;
    const _Float16* bp1 = wf + (size_t)o1 * 96 + half * 16;
    #pragma unroll
    for (int kk = 0; kk < 3; ++kk) {
        a[kk].q[0]  = *(const uint4*)(ap + kk * 32);        // K = half*16 + 0..7
        a[kk].q[1]  = *(const uint4*)(ap + kk * 32 + 8);    // K = half*16 + 8..15
        bL[kk].q[0] = *(const uint4*)(bp0 + kk * 32);       // K = half*16 + 0..7
        bL[kk].q[1] = *(const uint4*)(bp0 + kk * 32 + 8);   // K = half*16 + 8..15
        bH[kk].q[0] = *(const uint4*)(bp1 + kk * 32);
        bH[kk].q[1] = *(const uint4*)(bp1 + kk * 32 + 8);
    }

    v8f acc0 = {}, acc1 = {};
    #pragma unroll
    for (int kk = 0; kk < 3; ++kk) {
        acc0 = __builtin_amdgcn_wmma_f32_16x16x32_f16(
            false, a[kk].v, false, bL[kk].v, (short)0, acc0, false, false);
        acc1 = __builtin_amdgcn_wmma_f32_16x16x32_f16(
            false, a[kk].v, false, bH[kk].v, (short)0, acc1, false, false);
    }

    // ---- epilogue: +bias, two channels x 8 consecutive n -> 2x b128 each ----
    const float bs0 = bias[o0];
    const float bs1 = bias[o1];
    float* p0 = out + ((size_t)bb * COUT + o0) * NN + n0 + half * 8;
    float* p1 = out + ((size_t)bb * COUT + o1) * NN + n0 + half * 8;
    float4 v0a = { acc0[0] + bs0, acc0[1] + bs0, acc0[2] + bs0, acc0[3] + bs0 };
    float4 v0b = { acc0[4] + bs0, acc0[5] + bs0, acc0[6] + bs0, acc0[7] + bs0 };
    float4 v1a = { acc1[0] + bs1, acc1[1] + bs1, acc1[2] + bs1, acc1[3] + bs1 };
    float4 v1b = { acc1[4] + bs1, acc1[5] + bs1, acc1[6] + bs1, acc1[7] + bs1 };
    *(float4*)(p0)     = v0a;
    *(float4*)(p0 + 4) = v0b;
    *(float4*)(p1)     = v1a;
    *(float4*)(p1 + 4) = v1b;
}

// ---------------------------------------------------------------------------
// Kernel 5: per-channel sum / sumsq over B*N samples. One block per channel,
// fully coalesced reads, deterministic LDS tree reduction (no atomics).
// ---------------------------------------------------------------------------
__global__ __launch_bounds__(256) void stats_kernel(
    const float* __restrict__ out, float* __restrict__ stats)
{
    const int c = blockIdx.x;
    float s = 0.0f, s2 = 0.0f;
    for (int b = 0; b < BB; ++b) {
        const float* rowp = out + ((size_t)b * COUT + c) * NN;
        for (int n = threadIdx.x; n < NN; n += 256) {
            float v = rowp[n];
            s  += v;
            s2 += v * v;
        }
    }
    __shared__ float sh0[256];
    __shared__ float sh1[256];
    sh0[threadIdx.x] = s;
    sh1[threadIdx.x] = s2;
    __syncthreads();
    for (int st = 128; st > 0; st >>= 1) {
        if (threadIdx.x < st) {
            sh0[threadIdx.x] += sh0[threadIdx.x + st];
            sh1[threadIdx.x] += sh1[threadIdx.x + st];
        }
        __syncthreads();
    }
    if (threadIdx.x == 0) {
        stats[c]        = sh0[0];
        stats[COUT + c] = sh1[0];
    }
}

// ---------------------------------------------------------------------------
// Kernel 6: in-place BatchNorm + exact-erf GELU on (B, COUT, N).
// grid.y enumerates (b, c): channel constants are wave-uniform scalar loads.
// ---------------------------------------------------------------------------
__global__ __launch_bounds__(256) void finalize_kernel(
    float* __restrict__ out, const float* __restrict__ stats,
    const float* __restrict__ gamma, const float* __restrict__ beta)
{
    const int n = blockIdx.x * 256 + threadIdx.x;
    if (n >= NN) return;
    const int bc = blockIdx.y;          // b*COUT + c
    const int c  = bc % COUT;
    const float cnt = (float)(BB * NN);
    float mean = stats[c] / cnt;
    float var  = stats[COUT + c] / cnt - mean * mean;
    float inv  = rsqrtf(var + BN_EPS);
    float sc   = inv * gamma[c];
    float sh   = beta[c] - mean * sc;
    size_t idx = (size_t)bc * NN + n;
    float v = out[idx];
    v = v * sc + sh;
    v = 0.5f * v * (1.0f + erff(v * 0.70710678118654752f));  // exact GELU
    out[idx] = v;
}

// ---------------------------------------------------------------------------
extern "C" void kernel_launch(void* const* d_in, const int* in_sizes, int n_in,
                              void* d_out, int out_size, void* d_ws, size_t ws_size,
                              hipStream_t stream)
{
    const float* x     = (const float*)d_in[0];
    const int*   eidx  = (const int*)d_in[1];
    const float* w     = (const float*)d_in[2];
    const float* bias  = (const float*)d_in[3];
    const float* gamma = (const float*)d_in[4];
    const float* beta  = (const float*)d_in[5];
    float* out = (float*)d_out;

    // workspace layout (all offsets 256B-aligned)
    char* wsb = (char*)d_ws;
    const size_t XT_BYTES = (size_t)BB * NN * CIN * sizeof(_Float16);  // 15,360,000
    const size_t W_BYTES  = (size_t)COUT * 96 * sizeof(_Float16);      // 73,728
    _Float16* xt    = (_Float16*)(wsb);
    _Float16* rel   = (_Float16*)(wsb + XT_BYTES);
    _Float16* wf    = (_Float16*)(wsb + 2 * XT_BYTES);
    float*    stats = (float*)   (wsb + 2 * XT_BYTES + W_BYTES);

    // 1) transpose x -> xt (f16)
    {
        dim3 grid((NN + 31) / 32, (CIN + 31) / 32, BB);
        dim3 blk(32, 8, 1);
        transpose_to_f16<<<grid, blk, 0, stream>>>(x, xt);
    }
    // 2) weights f32 -> f16
    {
        int n = COUT * 96;
        wconv_kernel<<<(n + 255) / 256, 256, 0, stream>>>(w, wf, n);
    }
    // 3) max-relative gather/reduce -> rel (f16)
    {
        dim3 grid(NN / NPB, BB, 1);
        mrconv_kernel<<<grid, 192, 0, stream>>>(xt, eidx, rel);
    }
    // 4) grouped GEMM via WMMA, pre-BN result into d_out (B, COUT, N)
    {
        int totalWaves = M_TILES * GG * NT_PAIRS;                // 30000
        int blocks = totalWaves / WAVES_PER_BLOCK;               // 3750
        gemm_wmma_kernel<<<blocks, 32 * WAVES_PER_BLOCK, 0, stream>>>(
            xt, rel, wf, bias, out);
    }
    // 5) per-channel stats
    stats_kernel<<<COUT, 256, 0, stream>>>(out, stats);
    // 6) BN + GELU in place
    {
        dim3 grid((NN + 255) / 256, BB * COUT, 1);
        finalize_kernel<<<grid, 256, 0, stream>>>(out, stats, gamma, beta);
    }
}
